// SparseMultiHeadGATLayer_53300544143791
// MI455X (gfx1250) — matrix operand order repaired and checked by reading, not compiled, not run on hardware
//
#include <hip/hip_runtime.h>
#include <hip/hip_bf16.h>

typedef __attribute__((ext_vector_type(16))) __bf16 v16bf;
typedef __attribute__((ext_vector_type(8)))  float  v8f;

#define N_NODES 50000
#define N_EDGES 1600000
#define IN_F    256
#define OUT_F   64
#define N_HEADS 8
#define ALPHA   0.2f
#define EPS_F   1e-16f

__device__ __forceinline__ void atomAddF32(float* p, float v) {
  __hip_atomic_fetch_add(p, v, __ATOMIC_RELAXED, __HIP_MEMORY_SCOPE_AGENT);
}

__device__ __forceinline__ float lrelu(float x) {
  return x >= 0.0f ? x : ALPHA * x;
}

// ---------------------------------------------------------------------------
// Kernel 1: Wh[h][n][d] (bf16) = h[n][:] @ W[h][:][d], via v_wmma_f32_16x16x32_bf16.
// grid.x = N_NODES/16 M-tiles, 256 threads = 8 waves, wave w == head w.
// Each wave computes a 16(M) x 64(N) tile: 4 accumulators, K loop of 8 steps.
// ---------------------------------------------------------------------------
__global__ __launch_bounds__(256)
void gat_gemm_wmma(const float* __restrict__ h,
                   const float* __restrict__ W,
                   __bf16* __restrict__ Wh) {
  const int m0     = blockIdx.x * 16;
  const int head   = threadIdx.x >> 5;
  const int lane   = threadIdx.x & 31;
  const int l15    = lane & 15;
  const int hiHalf = lane >> 4;           // 0 or 1

  const float* __restrict__ hrow = h + (size_t)(m0 + l15) * IN_F;
  const float* __restrict__ Wb   = W + (size_t)head * IN_F * OUT_F;

  v8f acc[4] = {};

  for (int k0 = 0; k0 < IN_F; k0 += 32) {
    // A 16x32 bf16: lane<16 -> K {k0..k0+15}; lane>=16 -> +16
    const int ka = k0 + hiHalf * 16;
    v16bf A;
#pragma unroll
    for (int i = 0; i < 8; ++i) {
      A[i]     = (__bf16)hrow[ka + i];
      A[i + 8] = (__bf16)hrow[ka + 8 + i];
    }
    // B 32x16 bf16 per N-tile: lane = column, lane<16 -> K {k0..k0+15}, else +16
    const int kb = k0 + hiHalf * 16;
#pragma unroll
    for (int nt = 0; nt < 4; ++nt) {
      const int n = nt * 16 + l15;
      v16bf B;
#pragma unroll
      for (int i = 0; i < 16; ++i)
        B[i] = (__bf16)Wb[(size_t)(kb + i) * OUT_F + n];
      acc[nt] = __builtin_amdgcn_wmma_f32_16x16x32_bf16(
          false, A, false, B, (short)0, acc[nt], false, false);
    }
  }

  // D 16x16 f32: lane = column (l15), VGPR r -> M = hiHalf*8 + r
  __bf16* __restrict__ out = Wh + ((size_t)head * N_NODES + m0) * OUT_F;
#pragma unroll
  for (int nt = 0; nt < 4; ++nt) {
#pragma unroll
    for (int r = 0; r < 8; ++r) {
      const int M = hiHalf * 8 + r;
      out[(size_t)M * OUT_F + nt * 16 + l15] = (__bf16)acc[nt][r];
    }
  }
}

// ---------------------------------------------------------------------------
// Kernel 2: per-(head,node) attention logit halves (f32):
//   s_src[h][n] = Wh[h][n][:] . a[h][0:64],  s_dst[h][n] = Wh[h][n][:] . a[h][64:128]
// ---------------------------------------------------------------------------
__global__ __launch_bounds__(256)
void gat_scores(const __bf16* __restrict__ Wh,
                const float* __restrict__ a,
                float* __restrict__ s_src,
                float* __restrict__ s_dst) {
  const int tid = blockIdx.x * blockDim.x + threadIdx.x;
  if (tid >= N_HEADS * N_NODES) return;
  const int head = tid / N_NODES;

  const __bf16* __restrict__ w = Wh + (size_t)tid * OUT_F;
  const float*  __restrict__ as = a + (size_t)head * 2 * OUT_F;
  const float*  __restrict__ ad = as + OUT_F;

  float ss = 0.0f, sd = 0.0f;
#pragma unroll
  for (int d = 0; d < OUT_F; ++d) {
    const float wv = (float)w[d];
    ss += wv * as[d];
    sd += wv * ad[d];
  }
  s_src[tid] = ss;
  s_dst[tid] = sd;
}

// ---------------------------------------------------------------------------
// Kernel 3: edge pass 1 — e = leaky_relu(s_src[row] + s_dst[col]);
//   rowsum[h][row] += e   (reference quirk: segment SUM, not max)
// ---------------------------------------------------------------------------
__global__ __launch_bounds__(256)
void gat_edge_rowsum(const int* __restrict__ ei,
                     const float* __restrict__ s_src,
                     const float* __restrict__ s_dst,
                     float* __restrict__ rowsum) {
  const int e = blockIdx.x * blockDim.x + threadIdx.x;
  if (e >= N_EDGES) return;
  const int row = ei[e];
  const int col = ei[N_EDGES + e];
#pragma unroll
  for (int hd = 0; hd < N_HEADS; ++hd) {
    const int base = hd * N_NODES;
    const float ev = lrelu(s_src[base + row] + s_dst[base + col]);
    atomAddF32(&rowsum[base + row], ev);
  }
}

// ---------------------------------------------------------------------------
// Kernel 4: edge pass 2 — e_exp = exp(e - rowsum[row]); denom[h][row] += e_exp
// ---------------------------------------------------------------------------
__global__ __launch_bounds__(256)
void gat_edge_denom(const int* __restrict__ ei,
                    const float* __restrict__ s_src,
                    const float* __restrict__ s_dst,
                    const float* __restrict__ rowsum,
                    float* __restrict__ denom) {
  const int e = blockIdx.x * blockDim.x + threadIdx.x;
  if (e >= N_EDGES) return;
  const int row = ei[e];
  const int col = ei[N_EDGES + e];
#pragma unroll
  for (int hd = 0; hd < N_HEADS; ++hd) {
    const int base = hd * N_NODES;
    const float ev = lrelu(s_src[base + row] + s_dst[base + col]);
    const float ee = __expf(ev - rowsum[base + row]);
    atomAddF32(&denom[base + row], ee);
  }
}

// ---------------------------------------------------------------------------
// Kernel 5: edge pass 3 (the heavy one) —
//   att = e_exp / (denom[row] + eps);  out[row][h*64+d] += att * Wh[h][col][d]
// 16 lanes per (edge, head). STRIDED feature ownership: lane owns
// d = lane16 + 16*i, so each atomic/load issue covers one contiguous 64B
// region across the half-wave (one cacheline instead of four).
// ---------------------------------------------------------------------------
__global__ __launch_bounds__(256)
void gat_edge_aggregate(const int* __restrict__ ei,
                        const float* __restrict__ s_src,
                        const float* __restrict__ s_dst,
                        const float* __restrict__ rowsum,
                        const float* __restrict__ denom,
                        const __bf16* __restrict__ Wh,
                        float* __restrict__ out) {
  const long long tid = (long long)blockIdx.x * blockDim.x + threadIdx.x;
  const long long pair = tid >> 4;
  if (pair >= (long long)N_EDGES * N_HEADS) return;
  const int lane16 = (int)(tid & 15);
  const int e  = (int)(pair >> 3);
  const int hd = (int)(pair & 7);

  const int row = ei[e];
  const int col = ei[N_EDGES + e];
  const int base = hd * N_NODES;

  const float ev  = lrelu(s_src[base + row] + s_dst[base + col]);
  const float ee  = __expf(ev - rowsum[base + row]);
  const float att = ee / (denom[base + row] + EPS_F);

  const __bf16* __restrict__ w = Wh + ((size_t)base + col) * OUT_F;
  float* __restrict__ o =
      out + (size_t)row * (N_HEADS * OUT_F) + hd * OUT_F;

#pragma unroll
  for (int i = 0; i < 4; ++i) {
    const int d = lane16 + 16 * i;         // coalesced across the 16 lanes
    atomAddF32(&o[d], att * (float)w[d]);
  }
}

// ---------------------------------------------------------------------------
extern "C" void kernel_launch(void* const* d_in, const int* in_sizes, int n_in,
                              void* d_out, int out_size, void* d_ws, size_t ws_size,
                              hipStream_t stream) {
  const float* d_h  = (const float*)d_in[0];
  const float* d_W  = (const float*)d_in[1];
  const float* d_a  = (const float*)d_in[2];
  const int*   d_ei = (const int*)d_in[3];
  float* out = (float*)d_out;

  // Workspace layout (all offsets 256B-aligned)
  char* ws = (char*)d_ws;
  const size_t whBytes = (size_t)N_HEADS * N_NODES * OUT_F * sizeof(__bf16); // 51.2 MB
  const size_t svBytes = (size_t)N_HEADS * N_NODES * sizeof(float);          // 1.6 MB
  __bf16* Wh     = (__bf16*)(ws);
  float*  s_src  = (float*)(ws + whBytes);
  float*  s_dst  = (float*)(ws + whBytes + svBytes);
  float*  rowsum = (float*)(ws + whBytes + 2 * svBytes);
  float*  denom  = (float*)(ws + whBytes + 3 * svBytes);

  // Zero accumulators + output (we accumulate via atomics)
  hipMemsetAsync(rowsum, 0, 2 * svBytes, stream);                 // rowsum + denom
  hipMemsetAsync(out, 0, (size_t)out_size * sizeof(float), stream);

  // 1) WMMA GEMM: Wh = h @ W per head
  gat_gemm_wmma<<<N_NODES / 16, 256, 0, stream>>>(d_h, d_W, Wh);

  // 2) per-node attention score halves
  {
    const int total = N_HEADS * N_NODES;
    gat_scores<<<(total + 255) / 256, 256, 0, stream>>>(Wh, d_a, s_src, s_dst);
  }

  // 3) edge pass 1: segment-sum of leaky-relu logits
  gat_edge_rowsum<<<(N_EDGES + 255) / 256, 256, 0, stream>>>(d_ei, s_src, s_dst, rowsum);

  // 4) edge pass 2: exp + denominator
  gat_edge_denom<<<(N_EDGES + 255) / 256, 256, 0, stream>>>(d_ei, s_src, s_dst, rowsum, denom);

  // 5) edge pass 3: weighted scatter-aggregate
  {
    const long long total = (long long)N_EDGES * N_HEADS * 16;
    const int blocks = (int)((total + 255) / 256);
    gat_edge_aggregate<<<blocks, 256, 0, stream>>>(d_ei, s_src, s_dst, rowsum, denom, Wh, out);
  }
}
